// CausalMultiHeadSelfAttention_74311524155697
// MI455X (gfx1250) — compile-verified, matrix operand order checked by reading, not run on hardware
//
#include <hip/hip_runtime.h>

typedef _Float16 half_t;
typedef __attribute__((ext_vector_type(16))) _Float16 v16h;
typedef __attribute__((ext_vector_type(8)))  _Float16 v8h;
typedef __attribute__((ext_vector_type(8)))  float    v8f;

#define NB 2
#define NS 2048
#define ND 1024
#define NH 16
#define NDK 64
constexpr float THETA = 10000.0f;

// ---------------------------------------------------------------------------
// Fragment loader for V_WMMA_F32_16X16X32_F16 (wave32).
// A-matrix 16x32: lane l holds row m=l&15; halves h<8 -> K = (l>>4)*8 + h,
// halves h>=8 -> K = 16 + (l>>4)*8 + (h-8).  B-matrix (KxN) has the mirrored
// layout with n = l&15, same K-per-half mapping, i.e. identical loads from a
// matrix stored row-major along K (W[n][k], K[token][dim], Vt[dim][token]).
// Each fragment = two contiguous 16-byte loads per lane.
// ---------------------------------------------------------------------------
__device__ __forceinline__ v16h load_frag(const half_t* __restrict__ p,
                                          int ld, int row0, int k0) {
  const int lane = threadIdx.x & 31;
  const int m  = lane & 15;
  const int hi = lane >> 4;
  const half_t* base = p + (size_t)(row0 + m) * ld + k0 + hi * 8;
  v8h lo = *(const v8h*)(base);
  v8h hv = *(const v8h*)(base + 16);
  v16h r;
#pragma unroll
  for (int i = 0; i < 8; ++i) { r[i] = lo[i]; r[i + 8] = hv[i]; }
  return r;
}

__device__ __forceinline__ v8f wmma16(v16h a, v16h b, v8f c) {
  // (neg_a, A, neg_b, B, c_mod, C, reuse_a, reuse_b)
  return __builtin_amdgcn_wmma_f32_16x16x32_f16(false, a, false, b,
                                                (short)0, c, false, false);
}

// ---------------------------------------------------------------------------
// fp32 -> f16 conversion
// ---------------------------------------------------------------------------
__global__ void cvt_f32_f16_kernel(const float* __restrict__ src,
                                   half_t* __restrict__ dst, int n) {
  int i = blockIdx.x * blockDim.x + threadIdx.x;
  if (i < n) dst[i] = (half_t)src[i];
}

// ---------------------------------------------------------------------------
// QKV projection + RoPE.  grid = (B*S/32, H, 3), block = 32 (one wave).
// Each wave: 32 tokens x 64 dims (one head) of Q (z=0), K (z=1) or V (z=2).
// Per k-step: 2 A-frags + 4 B-frags feed 8 WMMAs (B reused across row tiles).
// Q,K stored [B,H,S,DK]; V stored transposed [B,H,DK,S].
// ---------------------------------------------------------------------------
__global__ void qkv_rope_kernel(const half_t* __restrict__ xh,
                                const half_t* __restrict__ Wqh,
                                const half_t* __restrict__ Wkh,
                                const half_t* __restrict__ Wvh,
                                const int* __restrict__ tpos,
                                half_t* __restrict__ Qh,
                                half_t* __restrict__ Kh,
                                half_t* __restrict__ Vth) {
  const int lane  = threadIdx.x & 31;
  const int nl    = lane & 15;
  const int hi    = lane >> 4;
  const int tile  = blockIdx.x;           // 32-row tile over B*S
  const int h     = blockIdx.y;           // head
  const int which = blockIdx.z;           // 0=Q 1=K 2=V
  const int rowBase = tile * 32;
  const int b     = rowBase / NS;
  const int sBase = rowBase - b * NS;

  const half_t* W = (which == 0) ? Wqh : (which == 1) ? Wkh : Wvh;

  v8f acc[2][4];
#pragma unroll
  for (int rt = 0; rt < 2; ++rt)
#pragma unroll
    for (int nt = 0; nt < 4; ++nt)
#pragma unroll
      for (int i = 0; i < 8; ++i) acc[rt][nt][i] = 0.0f;

  for (int k0 = 0; k0 < ND; k0 += 32) {
    v16h a0 = load_frag(xh, ND, rowBase,      k0);
    v16h a1 = load_frag(xh, ND, rowBase + 16, k0);
    __builtin_prefetch(xh + (size_t)(rowBase + (lane & 15)) * ND + k0 + 64, 0, 0);
#pragma unroll
    for (int nt = 0; nt < 4; ++nt) {
      v16h bf = load_frag(W, ND, h * NDK + nt * 16, k0);
      acc[0][nt] = wmma16(a0, bf, acc[0][nt]);
      acc[1][nt] = wmma16(a1, bf, acc[1][nt]);
    }
  }

  if (which < 2) {
    // RoPE: dim n pairs with n^1 -> adjacent lanes in C layout.
#pragma unroll
    for (int nt = 0; nt < 4; ++nt) {
      const int n  = nt * 16 + nl;            // dim within head, 0..63
      const int i2 = n & ~1;                  // 2*i
      const float inv_freq = __expf(-((float)i2 / (float)NDK) * __logf(THETA));
      const float sgn = (n & 1) ? 1.0f : -1.0f;
#pragma unroll
      for (int rt = 0; rt < 2; ++rt) {
#pragma unroll
        for (int r = 0; r < 8; ++r) {
          const int srow = sBase + rt * 16 + r + 8 * hi;
          const float pos = (float)tpos[b * NS + srow];
          const float ang = pos * inv_freq;
          const float c = __cosf(ang);
          const float s = __sinf(ang);
          const float v  = acc[rt][nt][r];
          const float pv = __shfl_xor(v, 1, 32);
          acc[rt][nt][r] = v * c + sgn * pv * s;
        }
      }
    }
  }

  if (which == 2) {
    // V transposed: Vth[((b*H+h)*DK + n)*S + s]; r-dim is contiguous in s.
#pragma unroll
    for (int rt = 0; rt < 2; ++rt) {
#pragma unroll
      for (int nt = 0; nt < 4; ++nt) {
        const int n = nt * 16 + nl;
        half_t* dst = Vth + ((size_t)(b * NH + h) * NDK + n) * NS
                          + sBase + rt * 16 + 8 * hi;
#pragma unroll
        for (int r = 0; r < 8; ++r) dst[r] = (half_t)acc[rt][nt][r];
      }
    }
  } else {
    half_t* out = (which == 0) ? Qh : Kh;
#pragma unroll
    for (int rt = 0; rt < 2; ++rt) {
#pragma unroll
      for (int nt = 0; nt < 4; ++nt) {
        const int n = nt * 16 + nl;
#pragma unroll
        for (int r = 0; r < 8; ++r) {
          const int srow = sBase + rt * 16 + r + 8 * hi;
          out[((size_t)(b * NH + h) * NS + srow) * NDK + n] = (half_t)acc[rt][nt][r];
        }
      }
    }
  }
}

// ---------------------------------------------------------------------------
// Flash attention. grid = (S/16, H, B), block = 32 (one wave).
// Per wave: 16 queries; iterate causal 32-key blocks with online softmax.
// ---------------------------------------------------------------------------
#define PSTRIDE 40   // halves; 80 bytes -> 16B aligned b128 LDS loads

__global__ void flash_attn_kernel(const half_t* __restrict__ Qh,
                                  const half_t* __restrict__ Kh,
                                  const half_t* __restrict__ Vth,
                                  half_t* __restrict__ AOh) {
  __shared__ half_t pTile[16 * PSTRIDE];

  const int lane = threadIdx.x & 31;
  const int nl   = lane & 15;
  const int hi   = lane >> 4;
  const int qt   = blockIdx.x;
  const int h    = blockIdx.y;
  const int b    = blockIdx.z;
  const int qBase = qt * 16;

  const half_t* Qhead = Qh  + (size_t)(b * NH + h) * NS * NDK;
  const half_t* Khead = Kh  + (size_t)(b * NH + h) * NS * NDK;
  const half_t* Vhead = Vth + (size_t)(b * NH + h) * NDK * NS;

  v16h qf0 = load_frag(Qhead, NDK, qBase, 0);
  v16h qf1 = load_frag(Qhead, NDK, qBase, 32);

  v8f o[4];
#pragma unroll
  for (int nt = 0; nt < 4; ++nt)
#pragma unroll
    for (int i = 0; i < 8; ++i) o[nt][i] = 0.0f;

  float mrun[8], lrun[8];
#pragma unroll
  for (int r = 0; r < 8; ++r) { mrun[r] = -1.0e30f; lrun[r] = 0.0f; }

  const float scale = 0.125f;   // 1/sqrt(64)
  const int kmax = qBase + 15;

  for (int kb = 0; kb <= kmax; kb += 32) {
    // ---- scores: 16q x 32k ------------------------------------------------
    v8f s0, s1;
#pragma unroll
    for (int i = 0; i < 8; ++i) { s0[i] = 0.0f; s1[i] = 0.0f; }
    {
      v16h kf;
      kf = load_frag(Khead, NDK, kb,      0);  s0 = wmma16(qf0, kf, s0);
      kf = load_frag(Khead, NDK, kb,      32); s0 = wmma16(qf1, kf, s0);
      kf = load_frag(Khead, NDK, kb + 16, 0);  s1 = wmma16(qf0, kf, s1);
      kf = load_frag(Khead, NDK, kb + 16, 32); s1 = wmma16(qf1, kf, s1);
    }

    // ---- scale, causal mask, online softmax ------------------------------
#pragma unroll
    for (int r = 0; r < 8; ++r) {
      const int q = qBase + r + 8 * hi;
      float e0 = (kb + nl      > q) ? -1.0e30f : s0[r] * scale;
      float e1 = (kb + 16 + nl > q) ? -1.0e30f : s1[r] * scale;

      float rm = fmaxf(e0, e1);
#pragma unroll
      for (int off = 1; off < 16; off <<= 1)
        rm = fmaxf(rm, __shfl_xor(rm, off, 32));

      const float mnew  = fmaxf(mrun[r], rm);
      const float alpha = __expf(mrun[r] - mnew);
      const float p0 = __expf(e0 - mnew);
      const float p1 = __expf(e1 - mnew);

      float rs = p0 + p1;
#pragma unroll
      for (int off = 1; off < 16; off <<= 1)
        rs += __shfl_xor(rs, off, 32);

      lrun[r] = lrun[r] * alpha + rs;
      mrun[r] = mnew;
#pragma unroll
      for (int nt = 0; nt < 4; ++nt) o[nt][r] *= alpha;

      const int row = r + 8 * hi;
      pTile[row * PSTRIDE + nl]      = (half_t)p0;
      pTile[row * PSTRIDE + 16 + nl] = (half_t)p1;
    }

    __syncthreads();                       // single wave: fence only
    v16h pf = load_frag(pTile, PSTRIDE, 0, 0);
    __syncthreads();

    // ---- O += P x V  (V is [DK,S] so B-frag is contiguous along keys) ----
#pragma unroll
    for (int nt = 0; nt < 4; ++nt) {
      v16h vf = load_frag(Vhead, NS, nt * 16, kb);
      o[nt] = wmma16(pf, vf, o[nt]);
    }
  }

  // ---- finalize + store f16 [B,S,D] --------------------------------------
#pragma unroll
  for (int nt = 0; nt < 4; ++nt) {
    const int dim = h * NDK + nt * 16 + nl;
#pragma unroll
    for (int r = 0; r < 8; ++r) {
      const int srow = qBase + r + 8 * hi;
      AOh[((size_t)(b * NS + srow)) * ND + dim] = (half_t)(o[nt][r] / lrun[r]);
    }
  }
}

// ---------------------------------------------------------------------------
// Output projection: out = AO @ Wo^T, fp32 result. grid=(B*S/32, D/64), blk=32.
// 32x64 tile per wave, same A/B reuse structure as the QKV GEMM.
// ---------------------------------------------------------------------------
__global__ void out_proj_kernel(const half_t* __restrict__ AOh,
                                const half_t* __restrict__ Woh,
                                float* __restrict__ out) {
  const int lane = threadIdx.x & 31;
  const int nl   = lane & 15;
  const int hi   = lane >> 4;
  const int rowBase = blockIdx.x * 32;
  const int colBase = blockIdx.y * NDK;

  v8f acc[2][4];
#pragma unroll
  for (int rt = 0; rt < 2; ++rt)
#pragma unroll
    for (int nt = 0; nt < 4; ++nt)
#pragma unroll
      for (int i = 0; i < 8; ++i) acc[rt][nt][i] = 0.0f;

  for (int k0 = 0; k0 < ND; k0 += 32) {
    v16h a0 = load_frag(AOh, ND, rowBase,      k0);
    v16h a1 = load_frag(AOh, ND, rowBase + 16, k0);
#pragma unroll
    for (int nt = 0; nt < 4; ++nt) {
      v16h bf = load_frag(Woh, ND, colBase + nt * 16, k0);
      acc[0][nt] = wmma16(a0, bf, acc[0][nt]);
      acc[1][nt] = wmma16(a1, bf, acc[1][nt]);
    }
  }

#pragma unroll
  for (int rt = 0; rt < 2; ++rt) {
#pragma unroll
    for (int nt = 0; nt < 4; ++nt) {
      const int col = colBase + nt * 16 + nl;
#pragma unroll
      for (int r = 0; r < 8; ++r) {
        const int row = rowBase + rt * 16 + r + 8 * hi;
        out[(size_t)row * ND + col] = acc[rt][nt][r];
      }
    }
  }
}

// ---------------------------------------------------------------------------
extern "C" void kernel_launch(void* const* d_in, const int* in_sizes, int n_in,
                              void* d_out, int out_size, void* d_ws, size_t ws_size,
                              hipStream_t stream) {
  const float* x  = (const float*)d_in[0];
  const int*   tp = (const int*)d_in[1];
  const float* Wq = (const float*)d_in[2];
  const float* Wk = (const float*)d_in[3];
  const float* Wv = (const float*)d_in[4];
  const float* Wo = (const float*)d_in[5];
  float* out = (float*)d_out;

  // Workspace carve-up (f16 buffers), 256B aligned.
  size_t off = 0;
  auto carve = [&](size_t bytes) {
    void* p = (char*)d_ws + off;
    off += (bytes + 255) & ~(size_t)255;
    return p;
  };
  const size_t nX = (size_t)NB * NS * ND;     // 4M
  const size_t nW = (size_t)ND * ND;          // 1M
  half_t* xh  = (half_t*)carve(nX * 2);
  half_t* Wqh = (half_t*)carve(nW * 2);
  half_t* Wkh = (half_t*)carve(nW * 2);
  half_t* Wvh = (half_t*)carve(nW * 2);
  half_t* Woh = (half_t*)carve(nW * 2);
  half_t* Qh  = (half_t*)carve(nX * 2);       // [B,H,S,DK]
  half_t* Kh  = (half_t*)carve(nX * 2);       // [B,H,S,DK]
  half_t* Vth = (half_t*)carve(nX * 2);       // [B,H,DK,S]
  half_t* AOh = (half_t*)carve(nX * 2);       // [B,S,D]
  (void)ws_size;

  // 1) fp32 -> f16 conversions
  cvt_f32_f16_kernel<<<(int)((nX + 255) / 256), 256, 0, stream>>>(x,  xh,  (int)nX);
  cvt_f32_f16_kernel<<<(int)((nW + 255) / 256), 256, 0, stream>>>(Wq, Wqh, (int)nW);
  cvt_f32_f16_kernel<<<(int)((nW + 255) / 256), 256, 0, stream>>>(Wk, Wkh, (int)nW);
  cvt_f32_f16_kernel<<<(int)((nW + 255) / 256), 256, 0, stream>>>(Wv, Wvh, (int)nW);
  cvt_f32_f16_kernel<<<(int)((nW + 255) / 256), 256, 0, stream>>>(Wo, Woh, (int)nW);

  // 2) QKV projections + RoPE (32-row tiles)
  qkv_rope_kernel<<<dim3(NB * NS / 32, NH, 3), 32, 0, stream>>>(
      xh, Wqh, Wkh, Wvh, tp, Qh, Kh, Vth);

  // 3) causal flash attention
  flash_attn_kernel<<<dim3(NS / 16, NH, NB), 32, 0, stream>>>(Qh, Kh, Vth, AOh);

  // 4) output projection (fp32 to d_out, 32-row tiles)
  out_proj_kernel<<<dim3(NB * NS / 32, ND / NDK), 32, 0, stream>>>(AOh, Woh, out);

  (void)in_sizes; (void)n_in; (void)out_size;
}
